// TransformerBlock_9887014716120
// MI455X (gfx1250) — compile-verified, matrix operand order checked by reading, not run on hardware
//
#include <hip/hip_runtime.h>
#include <hip/hip_bf16.h>
#include <math.h>

// ---------------------------------------------------------------------------
// Types for CDNA5 WMMA (wave32): v_wmma_f32_16x16x32_bf16
// ---------------------------------------------------------------------------
typedef __attribute__((ext_vector_type(16))) __bf16 v16bf;
typedef __attribute__((ext_vector_type(8)))  __bf16 v8bf;
typedef __attribute__((ext_vector_type(8)))  float  v8f;

// Problem constants (from the reference)
#define Bc   2
#define Sc   1024
#define Tc   1024
#define DIMc 2048
#define NHc  16
#define NKVc 4
#define HDc  128
#define FFNc 5632
#define DKV  (NKVc * HDc)   // 512

__device__ __forceinline__ v8f wmma_bf16(v16bf a, v16bf b, v8f c) {
  // 8 args: (neg_a, A, neg_b, B, c_mod, C, reuse_a, reuse_b)
  return __builtin_amdgcn_wmma_f32_16x16x32_bf16(false, a, false, b, (short)0, c,
                                                 false, false);
}

// 16 bf16 values along K, row fixed, contiguous memory (A or transposed-B frag).
// p -> &M[row][k0 + ((lane>=16)?8:0)]; element e holds
// K = (e&7) + (e>=8?16:0) + (lane>=16?8:0).  Two 16-byte (b128) loads.
__device__ __forceinline__ v16bf frag_load_bf16(const __bf16* __restrict__ p) {
  v8bf lo = *(const v8bf*)(p);
  v8bf hi = *(const v8bf*)(p + 16);
  return __builtin_shufflevector(lo, hi, 0, 1, 2, 3, 4, 5, 6, 7,
                                 8, 9, 10, 11, 12, 13, 14, 15);
}

// ---------------------------------------------------------------------------
// Weight convert + transpose: in f32 [R,C] -> out bf16 [C,R] (LDS 32x32 tile)
// ---------------------------------------------------------------------------
__global__ __launch_bounds__(256) void transpose_cvt(const float* __restrict__ in,
                                                     __bf16* __restrict__ out,
                                                     int R, int C) {
  __shared__ float t[32][33];
  int tx = threadIdx.x, ty = threadIdx.y;        // blockDim = (32, 8)
  int r0 = blockIdx.y * 32, c0 = blockIdx.x * 32;
#pragma unroll
  for (int i = 0; i < 4; ++i)
    t[ty + 8 * i][tx] = in[(size_t)(r0 + ty + 8 * i) * C + c0 + tx];
  __syncthreads();
#pragma unroll
  for (int i = 0; i < 4; ++i)
    out[(size_t)(c0 + ty + 8 * i) * R + r0 + tx] = (__bf16)t[tx][ty + 8 * i];
}

// V cache f32 [B,T,NKV,HD] -> bf16 V^T [B,NKV,HD,T]
__global__ __launch_bounds__(256) void vtrans_kernel(const float* __restrict__ vc,
                                                     __bf16* __restrict__ vT) {
  __shared__ float t[32][33];
  int tx = threadIdx.x, ty = threadIdx.y;        // blockDim = (32, 8)
  int bg = blockIdx.z;                           // b*NKV + g
  int b = bg >> 2, g = bg & 3;
  int t0 = blockIdx.x * 32, d0 = blockIdx.y * 32;
  const float* src = vc + (size_t)b * Tc * DKV + (size_t)g * HDc;
#pragma unroll
  for (int i = 0; i < 4; ++i)
    t[ty + 8 * i][tx] = src[(size_t)(t0 + ty + 8 * i) * DKV + d0 + tx];
  __syncthreads();
  __bf16* dst = vT + (size_t)bg * HDc * Tc;
#pragma unroll
  for (int i = 0; i < 4; ++i)
    dst[(size_t)(d0 + ty + 8 * i) * Tc + t0 + tx] = (__bf16)t[tx][ty + 8 * i];
}

// ---------------------------------------------------------------------------
// RMSNorm: one block per row, bf16 output
// ---------------------------------------------------------------------------
__global__ __launch_bounds__(256) void rmsnorm_kernel(const float* __restrict__ x,
                                                      const float* __restrict__ w,
                                                      __bf16* __restrict__ out,
                                                      int dim) {
  int row = blockIdx.x;
  const float* xr = x + (size_t)row * dim;
  float ss = 0.f;
  for (int i = threadIdx.x; i < dim; i += blockDim.x) {
    float v = xr[i];
    ss += v * v;
  }
  __shared__ float red[256];
  red[threadIdx.x] = ss;
  __syncthreads();
  for (int s = 128; s > 0; s >>= 1) {
    if ((int)threadIdx.x < s) red[threadIdx.x] += red[threadIdx.x + s];
    __syncthreads();
  }
  float rs = rsqrtf(red[0] / (float)dim + 1e-5f);
  __bf16* o = out + (size_t)row * dim;
  for (int i = threadIdx.x; i < dim; i += blockDim.x)
    o[i] = (__bf16)(xr[i] * rs * w[i]);
}

// ---------------------------------------------------------------------------
// NT WMMA GEMM: C[M,N] = A[M,K] * Bt[N,K]^T (+ Res), all-bf16 contiguous frags.
// Block 256 thr = 8 waves (4M x 2N); block tile 128x128; wave tile 32x64:
// 8 WMMAs per 12 b128 loads per k-step (A frag reused 4x, B frag 2x).
// ---------------------------------------------------------------------------
__global__ __launch_bounds__(256) void gemm_nt_wmma(const __bf16* __restrict__ A,
                                                    const __bf16* __restrict__ Bt,
                                                    const float* __restrict__ Res,
                                                    float* __restrict__ C,
                                                    int M, int N, int K) {
  int lane = threadIdx.x & 31;
  int w    = threadIdx.x >> 5;
  int wm   = w >> 1;                   // 0..3
  int wn   = w & 1;                    // 0..1
  int mB   = blockIdx.y * 128 + wm * 32;
  int nB   = blockIdx.x * 128 + wn * 64;
  int l15  = lane & 15;
  int lh8  = (lane >> 4) << 3;         // +8 K offset for lanes 16..31
  (void)M;

  v8f acc[2][4] = {};
  for (int k0 = 0; k0 < K; k0 += 32) {
    const __bf16* ap = A + (size_t)(mB + l15) * K + k0 + lh8;
    v16bf a0 = frag_load_bf16(ap);
    v16bf a1 = frag_load_bf16(ap + (size_t)16 * K);
    const __bf16* bp = Bt + (size_t)(nB + l15) * K + k0 + lh8;
    v16bf b0 = frag_load_bf16(bp);
    v16bf b1 = frag_load_bf16(bp + (size_t)16 * K);
    v16bf b2 = frag_load_bf16(bp + (size_t)32 * K);
    v16bf b3 = frag_load_bf16(bp + (size_t)48 * K);
    acc[0][0] = wmma_bf16(a0, b0, acc[0][0]);
    acc[0][1] = wmma_bf16(a0, b1, acc[0][1]);
    acc[0][2] = wmma_bf16(a0, b2, acc[0][2]);
    acc[0][3] = wmma_bf16(a0, b3, acc[0][3]);
    acc[1][0] = wmma_bf16(a1, b0, acc[1][0]);
    acc[1][1] = wmma_bf16(a1, b1, acc[1][1]);
    acc[1][2] = wmma_bf16(a1, b2, acc[1][2]);
    acc[1][3] = wmma_bf16(a1, b3, acc[1][3]);
  }
  int rbase = (lane >> 4) << 3;        // C layout: row = r + 8*(lane>=16)
  if (Res != nullptr) {                // uniform branch, hoisted out of loops
#pragma unroll
    for (int i = 0; i < 2; ++i)
#pragma unroll
      for (int j = 0; j < 4; ++j)
#pragma unroll
        for (int r = 0; r < 8; ++r) {
          size_t idx = (size_t)(mB + i * 16 + rbase + r) * N + nB + j * 16 + l15;
          C[idx] = acc[i][j][r] + Res[idx];
        }
  } else {
#pragma unroll
    for (int i = 0; i < 2; ++i)
#pragma unroll
      for (int j = 0; j < 4; ++j)
#pragma unroll
        for (int r = 0; r < 8; ++r) {
          size_t idx = (size_t)(mB + i * 16 + rbase + r) * N + nB + j * 16 + l15;
          C[idx] = acc[i][j][r];
        }
  }
}

// ---------------------------------------------------------------------------
// Fused gate/up NT GEMM + SiLU*mul epilogue -> bf16: out = silu(A*Wg^T)*(A*Wu^T)
// Block tile 64x128, wave tile 32x32 (x2 accumulator sets): 8 WMMA / 12 loads.
// ---------------------------------------------------------------------------
__global__ __launch_bounds__(256) void gateup_nt_wmma(const __bf16* __restrict__ A,
                                                      const __bf16* __restrict__ Wg,
                                                      const __bf16* __restrict__ Wu,
                                                      __bf16* __restrict__ Out,
                                                      int M, int N, int K) {
  int lane = threadIdx.x & 31;
  int w    = threadIdx.x >> 5;
  int wm   = w >> 2;
  int wn   = w & 3;
  int mB   = blockIdx.y * 64 + wm * 32;
  int nB   = blockIdx.x * 128 + wn * 32;
  int l15  = lane & 15;
  int lh8  = (lane >> 4) << 3;
  (void)M;

  v8f accg[2][2] = {};
  v8f accu[2][2] = {};
  for (int k0 = 0; k0 < K; k0 += 32) {
    const __bf16* ap = A + (size_t)(mB + l15) * K + k0 + lh8;
    v16bf a0 = frag_load_bf16(ap);
    v16bf a1 = frag_load_bf16(ap + (size_t)16 * K);
    const __bf16* gp = Wg + (size_t)(nB + l15) * K + k0 + lh8;
    v16bf g0 = frag_load_bf16(gp);
    v16bf g1 = frag_load_bf16(gp + (size_t)16 * K);
    const __bf16* up = Wu + (size_t)(nB + l15) * K + k0 + lh8;
    v16bf u0 = frag_load_bf16(up);
    v16bf u1 = frag_load_bf16(up + (size_t)16 * K);
    accg[0][0] = wmma_bf16(a0, g0, accg[0][0]);
    accg[0][1] = wmma_bf16(a0, g1, accg[0][1]);
    accg[1][0] = wmma_bf16(a1, g0, accg[1][0]);
    accg[1][1] = wmma_bf16(a1, g1, accg[1][1]);
    accu[0][0] = wmma_bf16(a0, u0, accu[0][0]);
    accu[0][1] = wmma_bf16(a0, u1, accu[0][1]);
    accu[1][0] = wmma_bf16(a1, u0, accu[1][0]);
    accu[1][1] = wmma_bf16(a1, u1, accu[1][1]);
  }
  int rbase = (lane >> 4) << 3;
#pragma unroll
  for (int i = 0; i < 2; ++i)
#pragma unroll
    for (int j = 0; j < 2; ++j)
#pragma unroll
      for (int r = 0; r < 8; ++r) {
        size_t idx = (size_t)(mB + i * 16 + rbase + r) * N + nB + j * 16 + l15;
        float g = accg[i][j][r];
        float u = accu[i][j][r];
        Out[idx] = (__bf16)((g / (1.0f + __expf(-g))) * u);
      }
}

// ---------------------------------------------------------------------------
// RoPE: rotate q (f32 in) -> qB bf16 shadow; rotate cached K in place (f32 out
// for d_out correctness) + kB bf16 shadow for the attention kernel.
// ---------------------------------------------------------------------------
__global__ __launch_bounds__(256) void rope_kernel(const float* __restrict__ q,
                                                   float* __restrict__ kc,
                                                   __bf16* __restrict__ qB,
                                                   __bf16* __restrict__ kB,
                                                   const float* __restrict__ freqs) {
  int idx = blockIdx.x * blockDim.x + threadIdx.x;
  const int total = Bc * Sc * (NHc + NKVc) * (HDc / 2);
  if (idx >= total) return;
  int d2   = idx & 63;
  int rest = idx >> 6;
  int head = rest % (NHc + NKVc);
  int tok  = rest / (NHc + NKVc);        // b*S + s
  int s    = tok & (Sc - 1);
  float c  = freqs[((size_t)s * (HDc / 2) + d2) * 2 + 0];
  float sn = freqs[((size_t)s * (HDc / 2) + d2) * 2 + 1];
  if (head < NHc) {
    size_t off = (size_t)tok * (NHc * HDc) + (size_t)head * HDc + 2 * d2;
    float x0 = q[off], x1 = q[off + 1];
    qB[off]     = (__bf16)(x0 * c - x1 * sn);
    qB[off + 1] = (__bf16)(x0 * sn + x1 * c);
  } else {
    size_t off = (size_t)tok * DKV + (size_t)(head - NHc) * HDc + 2 * d2;
    float x0 = kc[off], x1 = kc[off + 1];
    float r0 = x0 * c - x1 * sn;
    float r1 = x0 * sn + x1 * c;
    kc[off] = r0;  kc[off + 1] = r1;      // f32 cache output
    kB[off] = (__bf16)r0;  kB[off + 1] = (__bf16)r1;
  }
}

// ---------------------------------------------------------------------------
// Fused causal attention (flash-style online softmax), all-WMMA, bf16 operands.
// grid.x = B*NH (32), grid.y = S/128 (8); 8 waves/block, 16 q-rows per wave.
// ---------------------------------------------------------------------------
__global__ __launch_bounds__(256) void attn_kernel(const __bf16* __restrict__ qB,
                                                   const __bf16* __restrict__ kB,
                                                   const __bf16* __restrict__ vT,
                                                   __bf16* __restrict__ ctx) {
  int lane = threadIdx.x & 31;
  int w    = threadIdx.x >> 5;
  int bh   = blockIdx.x;
  int b    = bh >> 4;
  int h    = bh & 15;
  int g    = h >> 2;                       // NH/NKV = 4
  int m0   = blockIdx.y * 128 + w * 16;    // first q row for this wave
  int l15  = lane & 15;
  int lh8  = (lane >> 4) << 3;
  int rbase = lh8;                         // C-layout row offset

  const __bf16* qh = qB + (size_t)b * Sc * (NHc * HDc) + (size_t)h * HDc;
  const __bf16* kh = kB + (size_t)b * Tc * DKV + (size_t)g * HDc;
  const __bf16* vh = vT + (size_t)(b * NKVc + g) * HDc * Tc;  // [HD][T]

  // Preload q fragments for the full HD=128 (4 K-steps of 32)
  v16bf qf[4];
#pragma unroll
  for (int kk = 0; kk < 4; ++kk)
    qf[kk] = frag_load_bf16(qh + (size_t)(m0 + l15) * (NHc * HDc) + kk * 32 + lh8);

  v8f o[8] = {};                           // ctx accumulators over HD=128
  float mrow[8], lrow[8];
#pragma unroll
  for (int r = 0; r < 8; ++r) { mrow[r] = -1e30f; lrow[r] = 0.f; }

  // wave-private P staging tile (C-layout -> A-layout bounce; LDS in-order/wave)
  __shared__ __bf16 pb[8][16][40];

  const float scale = 0.08838834764831845f;   // 1/sqrt(128)
  const int tEnd = blockIdx.y * 128 + 128;    // causal: block-uniform bound

  for (int t0 = 0; t0 < tEnd; t0 += 32) {
    // ---- scores: two 16x16 tiles over t (q . K^T) ----
    v8f s01[2];
#pragma unroll
    for (int half = 0; half < 2; ++half) {
      int tt = t0 + 16 * half;
      v8f sa = {};
#pragma unroll
      for (int kk = 0; kk < 4; ++kk) {
        v16bf kf = frag_load_bf16(kh + (size_t)(tt + l15) * DKV + kk * 32 + lh8);
        sa = wmma_bf16(qf[kk], kf, sa);
      }
      s01[half] = sa;
    }
    // ---- online softmax (row stats across 16-lane halves via shfl_xor) ----
    int col0 = t0 + l15;
    int col1 = t0 + 16 + l15;
#pragma unroll
    for (int r = 0; r < 8; ++r) {
      int row = m0 + rbase + r;
      float v0 = s01[0][r] * scale + (col0 > row ? -1e9f : 0.f);
      float v1 = s01[1][r] * scale + (col1 > row ? -1e9f : 0.f);
      float mx = fmaxf(v0, v1);
#pragma unroll
      for (int off = 1; off < 16; off <<= 1) mx = fmaxf(mx, __shfl_xor(mx, off));
      float mnew  = fmaxf(mrow[r], mx);
      float alpha = __expf(mrow[r] - mnew);
      mrow[r] = mnew;
      float p0 = __expf(v0 - mnew);
      float p1 = __expf(v1 - mnew);
      float ps = p0 + p1;
#pragma unroll
      for (int off = 1; off < 16; off <<= 1) ps += __shfl_xor(ps, off);
      lrow[r] = lrow[r] * alpha + ps;
#pragma unroll
      for (int j = 0; j < 8; ++j) o[j][r] *= alpha;
      pb[w][rbase + r][l15]      = (__bf16)p0;
      pb[w][rbase + r][16 + l15] = (__bf16)p1;
    }
    // ---- reload P as A-fragment ----
    v16bf pf;
#pragma unroll
    for (int e = 0; e < 8; ++e) pf[e] = pb[w][l15][lh8 + e];
#pragma unroll
    for (int e = 0; e < 8; ++e) pf[e + 8] = pb[w][l15][lh8 + 16 + e];
    // ---- ctx += P(16x32) x V(32x128): V^T rows are contiguous in t ----
#pragma unroll
    for (int j = 0; j < 8; ++j) {
      v16bf vf = frag_load_bf16(vh + (size_t)(j * 16 + l15) * Tc + t0 + lh8);
      o[j] = wmma_bf16(pf, vf, o[j]);
    }
  }

  // ---- epilogue: ctx = o / l  (bf16 for the out-proj GEMM A operand) ----
  __bf16* cb = ctx + (size_t)b * Sc * (NHc * HDc) + (size_t)h * HDc;
#pragma unroll
  for (int r = 0; r < 8; ++r) {
    float inv = 1.0f / lrow[r];
    int row = m0 + rbase + r;
#pragma unroll
    for (int j = 0; j < 8; ++j)
      cb[(size_t)row * (NHc * HDc) + j * 16 + l15] = (__bf16)(o[j][r] * inv);
  }
}

// ---------------------------------------------------------------------------
// Host launcher
// ---------------------------------------------------------------------------
extern "C" void kernel_launch(void* const* d_in, const int* in_sizes, int n_in,
                              void* d_out, int out_size, void* d_ws, size_t ws_size,
                              hipStream_t stream) {
  (void)in_sizes; (void)n_in; (void)out_size; (void)ws_size;
  const float* x      = (const float*)d_in[0];
  const float* freqs  = (const float*)d_in[1];
  // d_in[2]/d_in[3]: zero caches (fully overwritten); d_in[4]: mask (causal inline)
  const float* wq     = (const float*)d_in[5];
  const float* wk     = (const float*)d_in[6];
  const float* wv     = (const float*)d_in[7];
  const float* wo     = (const float*)d_in[8];
  const float* wgate  = (const float*)d_in[9];
  const float* wup    = (const float*)d_in[10];
  const float* wdown  = (const float*)d_in[11];
  const float* anorm  = (const float*)d_in[12];
  const float* fnorm  = (const float*)d_in[13];

  float* out    = (float*)d_out;
  float* xo     = out;                                   // [B,S,DIM]
  float* ck_out = out + (size_t)Bc * Sc * DIMc;          // [1,B,T,NKV,HD]
  float* cv_out = ck_out + (size_t)Bc * Tc * DKV;        // [1,B,T,NKV,HD]

  // ---- workspace carve-up (bytes, 256-aligned chunks) ----
  char* p = (char*)d_ws;
  auto takeB = [&](size_t bytes) {
    char* r = p;
    p += (bytes + 255) & ~(size_t)255;
    return r;
  };
  __bf16* wqT = (__bf16*)takeB((size_t)DIMc * DIMc * 2);       // [2048,2048]
  __bf16* wkT = (__bf16*)takeB((size_t)DKV * DIMc * 2);        // [512,2048]
  __bf16* wvT = (__bf16*)takeB((size_t)DKV * DIMc * 2);
  __bf16* woT = (__bf16*)takeB((size_t)DIMc * DIMc * 2);
  __bf16* wgT = (__bf16*)takeB((size_t)FFNc * DIMc * 2);       // [5632,2048]
  __bf16* wuT = (__bf16*)takeB((size_t)FFNc * DIMc * 2);
  __bf16* wdT = (__bf16*)takeB((size_t)DIMc * FFNc * 2);       // [2048,5632]
  __bf16* h   = (__bf16*)takeB((size_t)2048 * DIMc * 2);       // norm out (bf16)
  float*  qf32= (float*) takeB((size_t)2048 * DIMc * 4);       // q pre-rope
  __bf16* qB  = (__bf16*)takeB((size_t)2048 * DIMc * 2);       // q post-rope bf16
  __bf16* kB  = (__bf16*)takeB((size_t)2048 * DKV * 2);        // K post-rope bf16
  __bf16* vT  = (__bf16*)takeB((size_t)Bc * NKVc * HDc * Tc * 2); // V^T bf16
  __bf16* ctx = (__bf16*)takeB((size_t)2048 * DIMc * 2);       // attn out bf16
  float*  x1  = (float*) takeB((size_t)2048 * DIMc * 4);       // post-attn resid
  __bf16* act = (__bf16*)takeB((size_t)2048 * FFNc * 2);       // silu*up bf16

  const int M = Bc * Sc;               // 2048 tokens
  dim3 blk(256);
  dim3 tblk(32, 8);

  // 0) one-pass weight convert+transpose to bf16 [N,K] (all L2-resident after)
  transpose_cvt<<<dim3(DIMc / 32, DIMc / 32), tblk, 0, stream>>>(wq, wqT, DIMc, DIMc);
  transpose_cvt<<<dim3(DKV  / 32, DIMc / 32), tblk, 0, stream>>>(wk, wkT, DIMc, DKV);
  transpose_cvt<<<dim3(DKV  / 32, DIMc / 32), tblk, 0, stream>>>(wv, wvT, DIMc, DKV);
  transpose_cvt<<<dim3(DIMc / 32, DIMc / 32), tblk, 0, stream>>>(wo, woT, DIMc, DIMc);
  transpose_cvt<<<dim3(FFNc / 32, DIMc / 32), tblk, 0, stream>>>(wgate, wgT, DIMc, FFNc);
  transpose_cvt<<<dim3(FFNc / 32, DIMc / 32), tblk, 0, stream>>>(wup,   wuT, DIMc, FFNc);
  transpose_cvt<<<dim3(DIMc / 32, FFNc / 32), tblk, 0, stream>>>(wdown, wdT, FFNc, DIMc);

  // 1) attn rmsnorm -> bf16
  rmsnorm_kernel<<<M, blk, 0, stream>>>(x, anorm, h, DIMc);
  // 2) QKV projections (K/V straight into the d_out cache regions: S==T)
  gemm_nt_wmma<<<dim3(DIMc / 128, M / 128), blk, 0, stream>>>(h, wqT, nullptr, qf32, M, DIMc, DIMc);
  gemm_nt_wmma<<<dim3(DKV / 128, M / 128), blk, 0, stream>>>(h, wkT, nullptr, ck_out, M, DKV, DIMc);
  gemm_nt_wmma<<<dim3(DKV / 128, M / 128), blk, 0, stream>>>(h, wvT, nullptr, cv_out, M, DKV, DIMc);
  // 3) RoPE -> bf16 q/K shadows (+ f32 K cache rewrite), V^T bf16
  {
    const int total = Bc * Sc * (NHc + NKVc) * (HDc / 2);
    rope_kernel<<<(total + 255) / 256, blk, 0, stream>>>(qf32, ck_out, qB, kB, freqs);
  }
  vtrans_kernel<<<dim3(Tc / 32, HDc / 32, Bc * NKVc), tblk, 0, stream>>>(cv_out, vT);
  // 4) fused causal attention -> bf16 ctx
  attn_kernel<<<dim3(Bc * NHc, Sc / 128), blk, 0, stream>>>(qB, kB, vT, ctx);
  // 5) output projection + residual
  gemm_nt_wmma<<<dim3(DIMc / 128, M / 128), blk, 0, stream>>>(ctx, woT, x, x1, M, DIMc, DIMc);
  // 6) ffn rmsnorm (reuse h)
  rmsnorm_kernel<<<M, blk, 0, stream>>>(x1, fnorm, h, DIMc);
  // 7) fused gate/up + SiLU*mul -> bf16
  gateup_nt_wmma<<<dim3(FFNc / 128, M / 64), blk, 0, stream>>>(h, wgT, wuT, act, M, FFNc, DIMc);
  // 8) down projection + residual -> final x output
  gemm_nt_wmma<<<dim3(DIMc / 128, M / 128), blk, 0, stream>>>(act, wdT, x1, xo, M, DIMc, FFNc);
}
